// NaiveCallTrans_89696097010305
// MI455X (gfx1250) — matrix operand order repaired and checked by reading, not compiled, not run on hardware
//
#include <hip/hip_runtime.h>
#include <math.h>
#include <stdint.h>

// ---------------- problem constants (from reference) ----------------
static constexpr int NN = 50000;   // nodes
static constexpr int EE = 100000;  // edges per relation
static constexpr int HH = 8;       // heads
static constexpr int DD = 96;      // per-head dim
static constexpr int FF = 768;     // feature dim (H*D)
static constexpr int BB = 256;     // graphs
static constexpr int CC = 31;      // classes

// ---------------- WMMA types ----------------
typedef __attribute__((ext_vector_type(16))) __bf16 v16bf;
typedef __attribute__((ext_vector_type(8)))  float  v8f;

// ============================================================
// utility kernels
// ============================================================
__global__ void fill_f32(float* __restrict__ p, float v, size_t n) {
  size_t i = (size_t)blockIdx.x * blockDim.x + threadIdx.x;
  if (i < n) p[i] = v;
}

__global__ void f32_to_bf16(const float* __restrict__ in, __bf16* __restrict__ out, size_t n) {
  size_t i = (size_t)blockIdx.x * blockDim.x + threadIdx.x;
  if (i < n) out[i] = (__bf16)in[i];
}

// out[n*K + k] = (bf16) in[k*Nc + n]   (weight transpose, so GEMM B-tiles are contiguous)
__global__ void __launch_bounds__(256)
f32_to_bf16_transpose(const float* __restrict__ in, __bf16* __restrict__ out, int K, int Nc) {
  __shared__ float t[32][33];
  const int n0 = blockIdx.x * 32, k0 = blockIdx.y * 32;
  const int tx = threadIdx.x & 31, ty = threadIdx.x >> 5;
  for (int i = ty; i < 32; i += 8)
    t[i][tx] = in[(size_t)(k0 + i) * Nc + (n0 + tx)];
  __syncthreads();
  for (int i = ty; i < 32; i += 8)
    out[(size_t)(n0 + i) * K + (k0 + tx)] = (__bf16)t[tx][i];
}

__global__ void add_inplace(float* __restrict__ out, const float* __restrict__ h, size_t n) {
  size_t i = (size_t)blockIdx.x * blockDim.x + threadIdx.x;
  if (i < n) out[i] += h[i];
}

// ============================================================
// async global -> LDS copy (CDNA5, ASYNCcnt-tracked)
// ============================================================
__device__ __forceinline__ void async_copy_b128(unsigned lds_off, const void* gptr) {
  asm volatile("global_load_async_to_lds_b128 %0, %1, off"
               :: "v"(lds_off), "v"(gptr)
               : "memory");
}
__device__ __forceinline__ void wait_asynccnt0() {
  asm volatile("s_wait_asynccnt 0x0" ::: "memory");
}

// ============================================================
// WMMA bf16 GEMM:  C[M,Nc] = A[M,K] @ W[K,Nc] + bias[Nc]  (f32 accum)
// WT is W transposed: WT[Nc,K].  block = 256 threads (8 waves),
// tile = 128x128, K step 32, double-buffered async LDS staging.
// ============================================================
#define TM 128
#define TN 128
#define TK 32
#define LDP 8  // pad: (32+8)*2 = 80B row stride, keeps 16B alignment

__global__ void __launch_bounds__(256)
wmma_gemm_bf16(const __bf16* __restrict__ A, const __bf16* __restrict__ WT,
               const float* __restrict__ bias, float* __restrict__ C,
               int M, int Nc, int K) {
  __shared__ __align__(16) __bf16 As[2][TM][TK + LDP];
  __shared__ __align__(16) __bf16 Bs[2][TN][TK + LDP];  // Bs[n][k]

  const int m0   = blockIdx.x * TM;
  const int n0   = blockIdx.y * TN;
  const int tid  = threadIdx.x;
  const int lane = tid & 31;
  const int wv   = tid >> 5;
  const int lrow = lane & 15;
  const int kbA  = (lane < 16) ? 0 : 8;   // A frag: K 0..7/16..23 vs 8..15/24..31
  const int kbB  = (lane < 16) ? 0 : 8;   // B frag: K 0..7/16..23 vs 8..15/24..31

  // staging geometry: 128 rows x 32 bf16 = 8KB = 512 x 16B pieces; thread does 2/matrix
  const int prow = tid >> 2;         // 0..63
  const int pseg = (tid & 3) * 8;    // bf16 element offset 0,8,16,24 (16B pieces)

  v8f zero = {0.f,0.f,0.f,0.f,0.f,0.f,0.f,0.f};
  v8f acc[8];
#pragma unroll
  for (int j = 0; j < 8; ++j) acc[j] = zero;

  const int nk = K / TK;

  auto stage = [&](int kt, int buf) {
    const int k0 = kt * TK;
    int r0 = m0 + prow;      if (r0 >= M) r0 = M - 1;   // clamp: stays in-buffer
    int r1 = m0 + prow + 64; if (r1 >= M) r1 = M - 1;
    async_copy_b128((unsigned)(uintptr_t)&As[buf][prow][pseg],
                    A + (size_t)r0 * K + k0 + pseg);
    async_copy_b128((unsigned)(uintptr_t)&As[buf][prow + 64][pseg],
                    A + (size_t)r1 * K + k0 + pseg);
    async_copy_b128((unsigned)(uintptr_t)&Bs[buf][prow][pseg],
                    WT + (size_t)(n0 + prow) * K + k0 + pseg);
    async_copy_b128((unsigned)(uintptr_t)&Bs[buf][prow + 64][pseg],
                    WT + (size_t)(n0 + prow + 64) * K + k0 + pseg);
  };

  stage(0, 0);
  wait_asynccnt0();
  __syncthreads();

  for (int kt = 0; kt < nk; ++kt) {
    const int cur = kt & 1;
    if (kt + 1 < nk) stage(kt + 1, cur ^ 1);  // overlap next tile with compute

    // A fragment for this wave's 16-row strip
    union { uint4 u[2]; v16bf v; } av;
    av.u[0] = *(const uint4*)&As[cur][wv * 16 + lrow][kbA];
    av.u[1] = *(const uint4*)&As[cur][wv * 16 + lrow][kbA + 16];

    // preload ALL 8 B fragments, then burst the WMMAs back-to-back
    union { uint4 u[2]; v16bf v; } bv[8];
#pragma unroll
    for (int j = 0; j < 8; ++j) {
      bv[j].u[0] = *(const uint4*)&Bs[cur][j * 16 + lrow][kbB];
      bv[j].u[1] = *(const uint4*)&Bs[cur][j * 16 + lrow][kbB + 16];
    }
#pragma unroll
    for (int j = 0; j < 8; ++j)
      acc[j] = __builtin_amdgcn_wmma_f32_16x16x32_bf16(
          false, av.v, false, bv[j].v, (short)0, acc[j], false, false);

    wait_asynccnt0();
    __syncthreads();
  }

  // store D (f32 16x16 layout: VGPR v -> row v (+8 for high lanes), lane%16 -> col)
  const int rbase = m0 + wv * 16 + ((lane < 16) ? 0 : 8);
#pragma unroll
  for (int j = 0; j < 8; ++j) {
    const int col = n0 + j * 16 + lrow;
    const float bb = bias[col];
#pragma unroll
    for (int v = 0; v < 8; ++v) {
      const int row = rbase + v;
      if (row < M) C[(size_t)row * Nc + col] = acc[j][v] + bb;
    }
  }
}

// ============================================================
// edge pipeline
// ============================================================
__device__ __forceinline__ void atomicMaxFloat(float* addr, float val) {
  unsigned int* ua = (unsigned int*)addr;
  unsigned int old = *ua;
  while (__uint_as_float(old) < val) {
    unsigned int assumed = old;
    old = atomicCAS(ua, assumed, __float_as_uint(val));
    if (old == assumed) break;
  }
}

// one wave per edge: score[e,h] = attn[h,:] . leaky_relu(fs[src]+fd[dst]); segment max
__global__ void __launch_bounds__(256)
edge_score_kernel(const float* __restrict__ fs, const float* __restrict__ fd,
                  const float* __restrict__ attn, const int* __restrict__ src,
                  const int* __restrict__ dst, float* __restrict__ score,
                  float* __restrict__ segmax, int E) {
  const int e = blockIdx.x * 8 + (threadIdx.x >> 5);
  const int lane = threadIdx.x & 31;
  if (e >= E) return;
  const int s = src[e], d = dst[e];
  const float* ps = fs + (size_t)s * FF;
  const float* pd = fd + (size_t)d * FF;
  for (int h = 0; h < HH; ++h) {
    float acc = 0.f;
    for (int i = lane; i < DD; i += 32) {
      float x = ps[h * DD + i] + pd[h * DD + i];
      x = (x > 0.f) ? x : 0.2f * x;  // leaky_relu(0.2)
      acc += x * attn[h * DD + i];
    }
    for (int off = 16; off; off >>= 1) acc += __shfl_xor(acc, off, 32);
    if (lane == 0) {
      score[(size_t)e * HH + h] = acc;
      atomicMaxFloat(&segmax[(size_t)d * HH + h], acc);
    }
  }
}

// score -> exp(score - max); den[dst] += ex
__global__ void edge_exp_kernel(float* __restrict__ score, const float* __restrict__ segmax,
                                float* __restrict__ den, const int* __restrict__ dst, int E) {
  const size_t i = (size_t)blockIdx.x * blockDim.x + threadIdx.x;
  if (i >= (size_t)E * HH) return;
  const int e = (int)(i >> 3), h = (int)(i & 7);
  const int d = dst[e];
  const float ex = expf(score[i] - segmax[(size_t)d * HH + h]);
  score[i] = ex;
  atomicAdd(&den[(size_t)d * HH + h], ex);
}

// out[dst] += fs[src] * (ex/den[dst]) ; one wave per edge
__global__ void __launch_bounds__(256)
edge_accum_kernel(const float* __restrict__ fs, const float* __restrict__ score,
                  const float* __restrict__ den, const int* __restrict__ src,
                  const int* __restrict__ dst, float* __restrict__ out, int E) {
  const int e = blockIdx.x * 8 + (threadIdx.x >> 5);
  const int lane = threadIdx.x & 31;
  if (e >= E) return;
  const int s = src[e], d = dst[e];
  for (int i = lane; i < FF; i += 32) {
    const int h = i / DD;
    const float alpha = score[(size_t)e * HH + h] / den[(size_t)d * HH + h];
    atomicAdd(&out[(size_t)d * FF + i], fs[(size_t)s * FF + i] * alpha);
  }
}

// ============================================================
// classifier head
// ============================================================
__global__ void __launch_bounds__(256)
fc1_kernel(const float* __restrict__ h, const int* __restrict__ gidx,
           const float* __restrict__ w1, const float* __restrict__ b1,
           float* __restrict__ hid) {
  const int g = blockIdx.x;
  const int j = threadIdx.x;  // 256 threads = hidden dim
  __shared__ float emb[FF];
  const int node = gidx[g];
  for (int i = j; i < FF; i += 256) emb[i] = h[(size_t)node * FF + i];
  __syncthreads();
  float a = b1[j];
  for (int k = 0; k < FF; ++k) a += emb[k] * w1[k * 256 + j];
  hid[g * 256 + j] = fmaxf(a, 0.f);
}

// one wave per graph: fc2 + log_softmax + NLL accumulate into out[0]
__global__ void __launch_bounds__(32)
head_kernel(const float* __restrict__ hid, const float* __restrict__ w2,
            const float* __restrict__ b2, const int* __restrict__ labels,
            float* __restrict__ out, int Bn, int C) {
  const int g = blockIdx.x;
  const int lane = threadIdx.x;
  float logit = -__builtin_inff();
  if (lane < C) {
    float a = b2[lane];
    for (int k = 0; k < 256; ++k) a += hid[g * 256 + k] * w2[k * C + lane];
    logit = a;
    out[1 + g * C + lane] = a;  // logits output
  }
  float mx = logit;
  for (int off = 16; off; off >>= 1) mx = fmaxf(mx, __shfl_xor(mx, off, 32));
  float ex = (lane < C) ? expf(logit - mx) : 0.f;
  float sm = ex;
  for (int off = 16; off; off >>= 1) sm += __shfl_xor(sm, off, 32);
  const float lse = mx + logf(sm);
  if (lane < C && lane == labels[g])
    atomicAdd(&out[0], -(logit - lse) / (float)Bn);
}

// ============================================================
// host orchestration
// ============================================================
extern "C" void kernel_launch(void* const* d_in, const int* in_sizes, int n_in,
                              void* d_out, int out_size, void* d_ws, size_t ws_size,
                              hipStream_t stream) {
  (void)in_sizes; (void)n_in; (void)out_size; (void)ws_size;

  const float* feats    = (const float*)d_in[0];
  const float* w_src[2] = { (const float*)d_in[1], (const float*)d_in[6] };
  const float* b_src[2] = { (const float*)d_in[2], (const float*)d_in[7] };
  const float* w_dst[2] = { (const float*)d_in[3], (const float*)d_in[8] };
  const float* b_dst[2] = { (const float*)d_in[4], (const float*)d_in[9] };
  const float* attnp[2] = { (const float*)d_in[5], (const float*)d_in[10] };
  const float* fc1w = (const float*)d_in[11];
  const float* fc1b = (const float*)d_in[12];
  const float* fc2w = (const float*)d_in[13];
  const float* fc2b = (const float*)d_in[14];
  const int* esrc[2] = { (const int*)d_in[15], (const int*)d_in[17] };
  const int* edst[2] = { (const int*)d_in[16], (const int*)d_in[18] };
  const int* gidx   = (const int*)d_in[19];
  const int* labels = (const int*)d_in[20];
  float* out = (float*)d_out;

  // ---- carve workspace ----
  char* p = (char*)d_ws;
  auto carve = [&](size_t bytes) -> char* {
    char* r = p; p += (bytes + 255) & ~(size_t)255; return r;
  };
  const size_t NF = (size_t)NN * FF;
  float*  hA     = (float*)carve(NF * 4);             // layer0 out / layer1 in
  float*  hB     = (float*)carve(NF * 4);             // layer1 out (final h)
  float*  fs     = (float*)carve(NF * 4);
  float*  fd     = (float*)carve(NF * 4);
  __bf16* hbf    = (__bf16*)carve(NF * 2);
  __bf16* wbfT   = (__bf16*)carve((size_t)FF * FF * 2);  // transposed bf16 weight
  float*  score  = (float*)carve((size_t)EE * HH * 4);
  float*  segmax = (float*)carve((size_t)NN * HH * 4);
  float*  den    = (float*)carve((size_t)NN * HH * 4);
  float*  hid    = (float*)carve((size_t)BB * 256 * 4);

  const int TPB = 256;
  auto gsz = [&](size_t n) { return (unsigned)((n + TPB - 1) / TPB); };
  const dim3 ggrid((NN + TM - 1) / TM, FF / TN);
  const dim3 tgrid(FF / 32, FF / 32);
  const unsigned eblocks = (EE + 7) / 8;
  const size_t WW = (size_t)FF * FF;

  for (int l = 0; l < 2; ++l) {
    const float* hsrc = (l == 0) ? feats : hA;
    float* ob = (l == 0) ? hA : hB;

    fill_f32<<<gsz(NF), TPB, 0, stream>>>(ob, 0.f, NF);
    f32_to_bf16<<<gsz(NF), TPB, 0, stream>>>(hsrc, hbf, NF);

    for (int r = 0; r < 2; ++r) {
      // fs = h @ Wsrc + bsrc
      f32_to_bf16_transpose<<<tgrid, TPB, 0, stream>>>(w_src[r] + (size_t)l * WW, wbfT, FF, FF);
      wmma_gemm_bf16<<<ggrid, 256, 0, stream>>>(hbf, wbfT, b_src[r] + l * FF, fs, NN, FF, FF);
      // fd = h @ Wdst + bdst
      f32_to_bf16_transpose<<<tgrid, TPB, 0, stream>>>(w_dst[r] + (size_t)l * WW, wbfT, FF, FF);
      wmma_gemm_bf16<<<ggrid, 256, 0, stream>>>(hbf, wbfT, b_dst[r] + l * FF, fd, NN, FF, FF);

      // edge softmax + aggregate
      fill_f32<<<gsz((size_t)NN * HH), TPB, 0, stream>>>(segmax, -__builtin_inff(), (size_t)NN * HH);
      fill_f32<<<gsz((size_t)NN * HH), TPB, 0, stream>>>(den, 0.f, (size_t)NN * HH);
      edge_score_kernel<<<eblocks, 256, 0, stream>>>(fs, fd, attnp[r] + (size_t)l * HH * DD,
                                                     esrc[r], edst[r], score, segmax, EE);
      edge_exp_kernel<<<gsz((size_t)EE * HH), TPB, 0, stream>>>(score, segmax, den, edst[r], EE);
      edge_accum_kernel<<<eblocks, 256, 0, stream>>>(fs, score, den, esrc[r], edst[r], ob, EE);
    }
    // residual
    add_inplace<<<gsz(NF), TPB, 0, stream>>>(ob, hsrc, NF);
  }

  // classifier head
  fc1_kernel<<<BB, 256, 0, stream>>>(hB, gidx, fc1w, fc1b, hid);
  fill_f32<<<1, 1, 0, stream>>>(out, 0.f, 1);
  head_kernel<<<BB, 32, 0, stream>>>(hid, fc2w, fc2b, labels, out, BB, CC);
}
